// Decoder_82059645157847
// MI455X (gfx1250) — matrix operand order repaired and checked by reading, not compiled, not run on hardware
//
#include <hip/hip_runtime.h>
#include <hip/hip_bf16.h>

// ---------------------------------------------------------------------------
// Decoder (embedding -> LSTM(64 steps) -> vocab projection) for MI455X gfx1250
// All GEMMs run as bf16 x bf16 -> f32 via v_wmma_f32_16x16x32_bf16 (wave32).
// GEMM staging uses GLOBAL_LOAD_ASYNC_TO_LDS_B128 (ASYNCcnt-tracked) when the
// toolchain declares the builtin.
// ---------------------------------------------------------------------------

#define BB    64          // batch
#define SS    64          // seq len
#define EE    256         // embed
#define HH    512         // hidden
#define VV    32000       // vocab
#define KIN   (EE + HH)   // 768
#define K2H   (2 * HH)    // 1024
#define G4H   (4 * HH)    // 2048
#define MM    (BB * SS)   // 4096 rows

typedef __bf16 bf8v  __attribute__((ext_vector_type(8)));
typedef __bf16 bf16v __attribute__((ext_vector_type(16)));
typedef float  f8v   __attribute__((ext_vector_type(8)));

#if defined(__HIP_DEVICE_COMPILE__) && __has_builtin(__builtin_amdgcn_global_load_async_to_lds_b128)
#define HAVE_ASYNC_LDS 1
#else
#define HAVE_ASYNC_LDS 0
#endif

// ---- helpers ---------------------------------------------------------------

__device__ __forceinline__ __bf16 f2bf(float f) {
    unsigned u = __builtin_bit_cast(unsigned, f);
    unsigned r = (u + 0x7FFFu + ((u >> 16) & 1u)) >> 16;   // round-nearest-even
    unsigned short s = (unsigned short)r;
    return __builtin_bit_cast(__bf16, s);
}

__device__ __forceinline__ bf16v cat8(bf8v lo, bf8v hi) {
    return __builtin_shufflevector(lo, hi, 0,1,2,3,4,5,6,7,8,9,10,11,12,13,14,15);
}

// 16-bit A-matrix 16x32 fragment (ISA 7.12.2): lanes 0-15 rows M=0..15 K 0..7 /
// 16..23; lanes 16-31 same rows, K 8..15 / 24..31.  rowp points at (row, k0).
__device__ __forceinline__ bf16v load_frag_a(const __bf16* rowp, int lane) {
    int kb = (lane >> 4) << 3;                    // 0 or 8
    bf8v lo = *(const bf8v*)(rowp + kb);
    bf8v hi = *(const bf8v*)(rowp + kb + 16);
    return cat8(lo, hi);
}

// 16-bit B-matrix 32x16 fragment: same K order as A (K 0..7/16..23 | 8..15/24..31).
__device__ __forceinline__ bf16v load_frag_b(const __bf16* rowp, int lane) {
    int kb = (lane >> 4) << 3;                    // 0 or 8
    bf8v lo = *(const bf8v*)(rowp + kb);
    bf8v hi = *(const bf8v*)(rowp + kb + 16);
    return cat8(lo, hi);
}

__device__ __forceinline__ f8v wmma_bf16(bf16v a, bf16v b, f8v c) {
    return __builtin_amdgcn_wmma_f32_16x16x32_bf16(false, a, false, b,
                                                   (short)0, c, false, false);
}

__device__ __forceinline__ float sigmoidf_fast(float x) {
    return 1.0f / (1.0f + __expf(-x));
}

#if HAVE_ASYNC_LDS
#define AS1 __attribute__((address_space(1)))
#define AS3 __attribute__((address_space(3)))
typedef int i4v __attribute__((vector_size(16)));   // matches builtin prototype
// direct global -> LDS async copy, 16B per lane, tracked by ASYNCcnt
__device__ __forceinline__ void async_cp16(void* lds, const void* gl) {
    __builtin_amdgcn_global_load_async_to_lds_b128((AS1 i4v*)gl, (AS3 i4v*)lds,
                                                   0, 0);
}
__device__ __forceinline__ void async_wait0() {
#if __has_builtin(__builtin_amdgcn_s_wait_asynccnt)
    __builtin_amdgcn_s_wait_asynccnt(0);
#else
    asm volatile("s_wait_asynccnt 0" ::: "memory");
#endif
}
#endif

// ---- kernel 1: f32 -> bf16 weight conversion -------------------------------

__global__ __launch_bounds__(256)
void cvt_f32_bf16(const float* __restrict__ x, __bf16* __restrict__ y, long n) {
    long i = (long)blockIdx.x * 256 + threadIdx.x;
    long stride = (long)gridDim.x * 256;
    for (; i < n; i += stride) y[i] = f2bf(x[i]);
}

// ---- kernel 2: embedding gather + context concat ----------------------------

__global__ __launch_bounds__(256)
void prep_inputs(const int* __restrict__ seq, const float* __restrict__ emb,
                 const float* __restrict__ ctx,
                 __bf16* __restrict__ lstm_in,   // [4096, 768]
                 __bf16* __restrict__ out_in) {  // [4096, 1024] (ctx half only)
    int r = blockIdx.x;                 // 0..4095  = b*S + s
    int b = r >> 6;
    int tok = seq[r];
    const float* er = emb + (long)tok * EE;
    const float* cr = ctx + (long)b * HH;
    __bf16* L = lstm_in + (long)r * KIN;
    __bf16* O = out_in  + (long)r * K2H;
    int t = threadIdx.x;
    L[t] = f2bf(er[t]);                                   // 256 embed cols
    for (int i = t; i < HH; i += 256) {                   // 512 context cols
        __bf16 v = f2bf(cr[i]);
        L[EE + i] = v;
        O[HH + i] = v;
    }
}

// ---- kernel 3: initial state ------------------------------------------------

__global__ __launch_bounds__(256)
void init_state(const float* __restrict__ h, const float* __restrict__ c,
                __bf16* __restrict__ hb, float* __restrict__ cw) {
    int i = blockIdx.x * 256 + threadIdx.x;   // 32768 = 64*512
    hb[i] = f2bf(h[i]);
    cw[i] = c[i];
}

// ---- kernel 4: WMMA GEMM  C[M,N] = A[M,K] * W[N,K]^T + bias (+bias2) --------
// Block: 256 threads (8 waves, 2x4), tile 64(M) x 256(N); wave tile 32x64
// (2 A-frags x 4 B-frags = 8 WMMA / k-step, B-frag reused across A-frags).

__global__ __launch_bounds__(256)
void gemm_bf16_nt(const __bf16* __restrict__ A, const __bf16* __restrict__ W,
                  const float* __restrict__ bias, const float* __restrict__ bias2,
                  float* __restrict__ C, int M, int N, int K) {
    __shared__ __align__(16) __bf16 As[64][40];    // +8 bf16 pad, keeps 16B align
    __shared__ __align__(16) __bf16 Bs[256][40];   // 25.6 KB total

    const int m0   = blockIdx.y * 64;
    const int n0   = blockIdx.x * 256;
    const int tid  = threadIdx.x;
    const int lane = tid & 31;
    const int wave = tid >> 5;
    const int mi   = wave & 1;        // 2 m-strips of 32
    const int ni   = wave >> 1;       // 4 n-strips of 64

    f8v acc[2][4] = {};

    const int ar = tid >> 2;                 // A stage: 64 rows x 4 segs of 8
    const int ac = (tid & 3) * 8;
    const int br = tid >> 2;                 // B stage: 4 blocks of 64 rows
    const int bc = (tid & 3) * 8;

    for (int kk = 0; kk < K; kk += 32) {
#if HAVE_ASYNC_LDS
        async_cp16(&As[ar][ac], A + (long)(m0 + ar) * K + kk + ac);
#pragma unroll
        for (int j = 0; j < 4; ++j)
            async_cp16(&Bs[br + 64 * j][bc],
                       W + (long)(n0 + br + 64 * j) * K + kk + bc);
        async_wait0();
#else
        *(bf8v*)&As[ar][ac] = *(const bf8v*)(A + (long)(m0 + ar) * K + kk + ac);
#pragma unroll
        for (int j = 0; j < 4; ++j)
            *(bf8v*)&Bs[br + 64 * j][bc] =
                *(const bf8v*)(W + (long)(n0 + br + 64 * j) * K + kk + bc);
        if (kk + 32 < K) {
            __builtin_prefetch(A + (long)(m0 + ar) * K + kk + 32 + ac, 0, 1);
            __builtin_prefetch(W + (long)(n0 + br) * K + kk + 32 + bc, 0, 1);
        }
#endif
        __syncthreads();

        bf16v a0 = load_frag_a(&As[mi * 32 + (lane & 15)][0], lane);
        bf16v a1 = load_frag_a(&As[mi * 32 + 16 + (lane & 15)][0], lane);
#pragma unroll
        for (int s = 0; s < 4; ++s) {
            int brow = ni * 64 + s * 16 + (lane & 15);
            bf16v bf = load_frag_b(&Bs[brow][0], lane);
            acc[0][s] = wmma_bf16(a0, bf, acc[0][s]);
            acc[1][s] = wmma_bf16(a1, bf, acc[1][s]);
        }
        __syncthreads();
    }

    // D layout: vgpr v -> row v (+8 for lanes 16-31), col = lane&15
#pragma unroll
    for (int u = 0; u < 2; ++u) {
        int mrow = m0 + mi * 32 + u * 16 + ((lane >> 4) << 3);
#pragma unroll
        for (int s = 0; s < 4; ++s) {
            int n = n0 + ni * 64 + s * 16 + (lane & 15);
            float bv = bias[n] + (bias2 ? bias2[n] : 0.0f);
#pragma unroll
            for (int v = 0; v < 8; ++v) {
                C[(long)(mrow + v) * N + n] = acc[u][s][v] + bv;
            }
        }
    }
}

// ---- kernel 5: one LSTM step (fused gates-GEMM + pointwise) -----------------
// gates[64,2048] = pre[:,t,:] + h @ w_hh^T ; 128 wave tiles of 16 batch rows x
// 16 hidden cols, each computing all four gate columns for pointwise fusion.

__global__ __launch_bounds__(256)
void lstm_step(const __bf16* __restrict__ hprev,  // [64, 512] bf16
               const __bf16* __restrict__ whh,    // [2048, 512] bf16
               const float*  __restrict__ pre,    // [4096, 2048] f32
               float*  __restrict__ c_ws,         // [64, 512] f32
               __bf16* __restrict__ hnext,        // [64, 512] bf16
               __bf16* __restrict__ out_in,       // [4096, 1024] bf16
               int t) {
    const int wave = threadIdx.x >> 5;
    const int lane = threadIdx.x & 31;
    const int tile = blockIdx.x * 8 + wave;   // 0..127
    const int mi   = tile & 3;                // batch subtile: rows mi*16..
    const int ji   = tile >> 2;               // 0..31: hidden cols ji*16..

    f8v acc[4] = {};

    const __bf16* arowp = hprev + (long)(mi * 16 + (lane & 15)) * HH;
    const int jcolw = ji * 16 + (lane & 15);

    for (int kk = 0; kk < HH; kk += 32) {
        bf16v afrag = load_frag_a(arowp + kk, lane);
#pragma unroll
        for (int g = 0; g < 4; ++g) {
            const __bf16* brp = whh + (long)(g * HH + jcolw) * HH + kk;
            acc[g] = wmma_bf16(afrag, load_frag_b(brp, lane), acc[g]);
        }
    }

    const int b0  = mi * 16 + ((lane >> 4) << 3);
    const int col = ji * 16 + (lane & 15);
#pragma unroll
    for (int v = 0; v < 8; ++v) {
        int  b   = b0 + v;
        long prr = ((long)b * SS + t) * G4H;
        float xi = acc[0][v] + pre[prr + 0 * HH + col];
        float xf = acc[1][v] + pre[prr + 1 * HH + col];
        float xg = acc[2][v] + pre[prr + 2 * HH + col];
        float xo = acc[3][v] + pre[prr + 3 * HH + col];
        float I = sigmoidf_fast(xi);
        float F = sigmoidf_fast(xf);
        float G = tanhf(xg);
        float O = sigmoidf_fast(xo);
        long ci = (long)b * HH + col;
        float cn = F * c_ws[ci] + I * G;
        c_ws[ci] = cn;
        float hn = O * tanhf(cn);
        __bf16 hb = f2bf(hn);
        hnext[ci] = hb;
        out_in[((long)b * SS + t) * K2H + col] = hb;   // hs half of out_in
    }
}

// ---- workspace layout (bytes, all 256-aligned) ------------------------------

#define OFF_WIH  0L                       // 2048*768*2  = 3,145,728
#define OFF_WHH  3145728L                 // 2048*512*2  = 2,097,152
#define OFF_WFC  5242880L                 // 32000*1024*2= 65,536,000
#define OFF_LIN  70778880L                // 4096*768*2  = 6,291,456
#define OFF_OIN  77070336L                // 4096*1024*2 = 8,388,608
#define OFF_PRE  85458944L                // 4096*2048*4 = 33,554,432
#define OFF_H0   119013376L               // 64*512*2    = 65,536
#define OFF_H1   119078912L               // 64*512*2    = 65,536
#define OFF_C    119144448L               // 64*512*4    = 131,072
// total ~113.8 MiB

extern "C" void kernel_launch(void* const* d_in, const int* in_sizes, int n_in,
                              void* d_out, int out_size, void* d_ws, size_t ws_size,
                              hipStream_t stream) {
    const int*   target_seq = (const int*)  d_in[0];
    const float* context    = (const float*)d_in[1];
    const float* h0         = (const float*)d_in[2];
    const float* c0         = (const float*)d_in[3];
    const float* emb_table  = (const float*)d_in[4];
    const float* w_ih       = (const float*)d_in[5];
    const float* w_hh       = (const float*)d_in[6];
    const float* b_ih       = (const float*)d_in[7];
    const float* b_hh       = (const float*)d_in[8];
    const float* w_fc       = (const float*)d_in[9];
    const float* b_fc       = (const float*)d_in[10];
    float* logits = (float*)d_out;

    char* ws = (char*)d_ws;
    __bf16* w_ih_b  = (__bf16*)(ws + OFF_WIH);
    __bf16* w_hh_b  = (__bf16*)(ws + OFF_WHH);
    __bf16* w_fc_b  = (__bf16*)(ws + OFF_WFC);
    __bf16* lstm_in = (__bf16*)(ws + OFF_LIN);
    __bf16* out_in  = (__bf16*)(ws + OFF_OIN);
    float*  pre     = (float*) (ws + OFF_PRE);
    __bf16* hbuf0   = (__bf16*)(ws + OFF_H0);
    __bf16* hbuf1   = (__bf16*)(ws + OFF_H1);
    float*  c_ws    = (float*) (ws + OFF_C);

    // 1. weight precision conversion (one pass over HBM)
    cvt_f32_bf16<<<2048, 256, 0, stream>>>(w_ih, w_ih_b, (long)G4H * KIN);
    cvt_f32_bf16<<<2048, 256, 0, stream>>>(w_hh, w_hh_b, (long)G4H * HH);
    cvt_f32_bf16<<<8192, 256, 0, stream>>>(w_fc, w_fc_b, (long)VV * K2H);

    // 2. embedding gather + context concat
    prep_inputs<<<MM, 256, 0, stream>>>(target_seq, emb_table, context,
                                        lstm_in, out_in);

    // 3. initial LSTM state
    init_state<<<(BB * HH) / 256, 256, 0, stream>>>(h0, c0, hbuf0, c_ws);

    // 4. pre-gates GEMM: pre = lstm_in @ w_ih^T + b_ih + b_hh   [4096 x 2048]
    gemm_bf16_nt<<<dim3(G4H / 256, MM / 64), 256, 0, stream>>>(
        lstm_in, w_ih_b, b_ih, b_hh, pre, MM, G4H, KIN);

    // 5. LSTM recurrence (64 fused WMMA step kernels, double-buffered h)
    for (int t = 0; t < SS; ++t) {
        __bf16* hp = (t & 1) ? hbuf1 : hbuf0;
        __bf16* hn = (t & 1) ? hbuf0 : hbuf1;
        lstm_step<<<16, 256, 0, stream>>>(hp, w_hh_b, pre, c_ws, hn, out_in, t);
    }

    // 6. vocab projection: logits = out_in @ w_fc^T + b_fc   [4096 x 32000]
    gemm_bf16_nt<<<dim3(VV / 256, MM / 64), 256, 0, stream>>>(
        out_in, w_fc_b, b_fc, nullptr, logits, MM, VV, K2H);
}
